// SequentialPredictorWithVelocity_11708080849549
// MI455X (gfx1250) — matrix operand order repaired and checked by reading, not compiled, not run on hardware
//
#include <hip/hip_runtime.h>
#include <hip/hip_bf16.h>

// ---------- CDNA5 wave32 / WMMA types ----------
typedef __attribute__((ext_vector_type(16))) _Float16 v16h;
typedef __attribute__((ext_vector_type(8)))  _Float16 v8h;
typedef __attribute__((ext_vector_type(8)))  float    v8f;

union AFrag { v16h v; v8h h8[2]; _Float16 e[16]; };

#define HDIM     32
#define T_IN     8
#define OUT_LEN  12
#define WAVES    4          // waves per block, 16 batch rows each
#define SCR_STRIDE 72       // halves per scratch row (16x64 tile + pad)

// ---- workspace layout: f16 B-fragments (each fragment = 32 lanes x 16 halves = 512 halves) ----
enum {
  WHHP_OFF = 0,        // 8 frags  (K=32,  N=128)
  WHHV_OFF = 4096,     // 8 frags
  W1_OFF   = 8192,     // 16 frags (K=128, N=64)
  W2_OFF   = 16384,    // 8 frags  (K=64,  N=64)
  W3_OFF   = 20480,
  W4_OFF   = 24576,
  W5_OFF   = 28672,    // 2 frags  (K=64,  N=2 padded to 16)
  WSB_HALVES = 29696
};
enum {  // fp32 region (floats), placed after the f16 region
  WIHP_F = 0, BP_F = 256, WIHV_F = 384, BV_F = 640,
  B1_F = 768, B2_F = 832, B3_F = 896, B4_F = 960, B5_F = 1024, WSF_FLOATS = 1040
};

// ---------- branchless transcendentals (gfx1250 TRANS pipe; co-executes with WMMA) ----------
#define LOG2E 1.4426950408889634f

__device__ __forceinline__ float fast_exp2(float x) { return __builtin_amdgcn_exp2f(x); }
__device__ __forceinline__ float fast_rcp(float x)  { return __builtin_amdgcn_rcpf(x); }

__device__ __forceinline__ float tanh_fast(float x) {
#if __has_builtin(__builtin_amdgcn_tanhf)
  return __builtin_amdgcn_tanhf(x);          // v_tanh_f32 (CDNA5 hardware TRANS op)
#else
  // tanh(x) = 1 - 2/(exp2(2x*log2e)+1); branchless, saturates at +-1
  return 1.0f - 2.0f * fast_rcp(1.0f + fast_exp2(2.0f * LOG2E * x));
#endif
}

__device__ __forceinline__ float sigf(float x) {
#if __has_builtin(__builtin_amdgcn_tanhf)
  // sigmoid(x) = 0.5 + 0.5*tanh(x/2): 1 TRANS + 2 VALU via hardware tanh
  return __builtin_fmaf(0.5f, __builtin_amdgcn_tanhf(0.5f * x), 0.5f);
#else
  // 1/(1+e^-x) = rcp(1 + exp2(-x*log2e)); branchless
  return fast_rcp(1.0f + fast_exp2(-LOG2E * x));
#endif
}

// ---------- device helpers ----------
__device__ __forceinline__ v8f wmma_f16(v16h a, v16h b, v8f c) {
  // D = A(16x32 f16) * B(32x16 f16) + C(16x16 f32)
  return __builtin_amdgcn_wmma_f32_16x16x32_f16(false, a, false, b, (short)0, c, false, false);
}

__device__ __forceinline__ v16h load_bfrag(const _Float16* __restrict__ wsB, int frag, int lane) {
  return *(const v16h*)(wsB + (size_t)frag * 512 + lane * 16);  // 32B aligned
}

// A-matrix 16x32 f16 layout: lane<16 holds M=lane, K={kOff+0..7, kOff+16..23};
// lane>=16 holds M=lane-16, K={kOff+8..15, kOff+24..31}
__device__ __forceinline__ v16h load_afrag(const _Float16* __restrict__ scr, int lane, int kOff) {
  const int m = lane & 15, hs = lane >> 4;
  const _Float16* p = scr + m * SCR_STRIDE + kOff + 8 * hs;
  AFrag a;
  a.h8[0] = *(const v8h*)(p);
  a.h8[1] = *(const v8h*)(p + 16);
  return a.v;
}

__device__ __forceinline__ void lds_fence() {
  // DS ops are in-order within a wave; wait for store data to land before dependent loads.
  __builtin_amdgcn_wave_barrier();
  asm volatile("s_wait_dscnt 0" ::: "memory");
  __builtin_amdgcn_wave_barrier();
}

// One LSTM cell for a 16-row batch tile.
// xt: per-lane copy of the 2-dim input of the 8 rows this lane owns (C-layout rows).
// cst: cell state in C-layout (2 N-tiles x 8). hA: hidden state as WMMA A-fragment (out).
__device__ __forceinline__ void lstm_cell(
    const float2 xt[8], float cst[16], v16h& hA,
    const _Float16* __restrict__ wsB, int whhBase,
    const float* __restrict__ wih, const float* __restrict__ bias,
    _Float16* __restrict__ scr, int lane)
{
  const int col = lane & 15, hs = lane >> 4;
#pragma unroll
  for (int t2 = 0; t2 < 2; ++t2) {          // hidden N-tile (H=32 -> 2 tiles)
    float g[4][8];
#pragma unroll
    for (int p = 0; p < 4; ++p) {           // i, f, g, o gate groups
      const int tile = p * 2 + t2;          // gates = [i(0..31) f(32..63) g(64..95) o(96..127)]
      v8f acc = {};
      acc = wmma_f16(hA, load_bfrag(wsB, whhBase + tile, lane), acc);
      const int n = tile * 16 + col;
      const float w0 = wih[2 * n], w1 = wih[2 * n + 1], bb = bias[n];
#pragma unroll
      for (int r = 0; r < 8; ++r)
        g[p][r] = acc[r] + bb + xt[r].x * w0 + xt[r].y * w1;
    }
#pragma unroll
    for (int r = 0; r < 8; ++r) {
      float c = sigf(g[1][r]) * cst[t2 * 8 + r] + sigf(g[0][r]) * tanh_fast(g[2][r]);
      cst[t2 * 8 + r] = c;
      float h = sigf(g[3][r]) * tanh_fast(c);
      scr[(r + 8 * hs) * SCR_STRIDE + t2 * 16 + col] = (_Float16)h;
    }
  }
  lds_fence();
  hA = load_afrag(scr, lane, 0);
}

// Dense layer: out(16xNTILES*16) = A(16x32*KCH) @ W + b, optional leaky-relu, result -> LDS scratch.
template<int KCH, int APPLY_LRELU, int NTILES>
__device__ __forceinline__ void mlp_layer(
    const v16h A[KCH], const _Float16* __restrict__ wsB, int wBase,
    const float* __restrict__ bias, _Float16* __restrict__ scr, int lane)
{
  const int col = lane & 15, hs = lane >> 4;
#pragma unroll
  for (int tile = 0; tile < NTILES; ++tile) {
    v8f acc = {};
#pragma unroll
    for (int k = 0; k < KCH; ++k)
      acc = wmma_f16(A[k], load_bfrag(wsB, wBase + k * NTILES + tile, lane), acc);
    const float bb = bias[tile * 16 + col];
#pragma unroll
    for (int r = 0; r < 8; ++r) {
      float v = acc[r] + bb;
      if (APPLY_LRELU) v = (v >= 0.0f) ? v : 0.1f * v;
      scr[(r + 8 * hs) * SCR_STRIDE + tile * 16 + col] = (_Float16)v;
    }
  }
  lds_fence();
}

__device__ __forceinline__ void head_step(
    int step, float* __restrict__ out, int rowBase,
    v16h hA_p, v16h& hA_v, float c_v[16],
    const v16h noiseA[2],
    float2 last_[8], const float2 x0_[8], const float2 xm4_[8],
    const _Float16* __restrict__ wsB, const float* __restrict__ wsF,
    _Float16* __restrict__ scr, int lane)
{
  const int col = lane & 15, hs = lane >> 4;
  // velocity input v = (last - x0)/T
  float2 v[8];
#pragma unroll
  for (int r = 0; r < 8; ++r) {
    v[r].x = (last_[r].x - x0_[r].x) * (1.0f / (float)T_IN);
    v[r].y = (last_[r].y - x0_[r].y) * (1.0f / (float)T_IN);
  }
  lstm_cell(v, c_v, hA_v, wsB, WHHV_OFF / 512, wsF + WIHV_F, wsF + BV_F, scr, lane);

  // y = cat([hp, hv, noise])  -> K = 128 = 4 chunks of 32
  v16h A4[4] = { hA_p, hA_v, noiseA[0], noiseA[1] };
  mlp_layer<4, 1, 4>(A4, wsB, W1_OFF / 512, wsF + B1_F, scr, lane);   // lrelu(W1)
  v16h A2[2] = { load_afrag(scr, lane, 0), load_afrag(scr, lane, 32) };
  mlp_layer<2, 0, 4>(A2, wsB, W2_OFF / 512, wsF + B2_F, scr, lane);   // W2 (no lrelu)
  A2[0] = load_afrag(scr, lane, 0); A2[1] = load_afrag(scr, lane, 32);
  mlp_layer<2, 1, 4>(A2, wsB, W3_OFF / 512, wsF + B3_F, scr, lane);   // lrelu(W3)
  A2[0] = load_afrag(scr, lane, 0); A2[1] = load_afrag(scr, lane, 32);
  mlp_layer<2, 1, 4>(A2, wsB, W4_OFF / 512, wsF + B4_F, scr, lane);   // lrelu(W4)
  A2[0] = load_afrag(scr, lane, 0); A2[1] = load_afrag(scr, lane, 32);

  // W5: 64 -> 2 (N padded to one 16-wide tile; cols >= 2 are zero weights)
  v8f acc = {};
  acc = wmma_f16(A2[0], load_bfrag(wsB, W5_OFF / 512 + 0, lane), acc);
  acc = wmma_f16(A2[1], load_bfrag(wsB, W5_OFF / 512 + 1, lane), acc);

  const float bb = (col == 0) ? wsF[B5_F] : wsF[B5_F + 1];
  float yfin[8];
#pragma unroll
  for (int r = 0; r < 8; ++r) {
    const float res = (col == 0) ? (last_[r].x - xm4_[r].x) * (1.0f / 3.0f)
                                 : (last_[r].y - xm4_[r].y) * (1.0f / 3.0f);
    yfin[r] = acc[r] + bb + res;
  }
  if (col < 2) {
#pragma unroll
    for (int r = 0; r < 8; ++r)
      out[(size_t)(rowBase + r + 8 * hs) * (OUT_LEN * 2) + step * 2 + col] = yfin[r];
  }
  // broadcast y columns (held in lanes 0,1 / 16,17) to all lanes; last += y
#pragma unroll
  for (int r = 0; r < 8; ++r) {
    float y0 = __shfl(yfin[r], (lane & 16) + 0, 32);
    float y1 = __shfl(yfin[r], (lane & 16) + 1, 32);
    last_[r].x += y0;
    last_[r].y += y1;
  }
}

// ---------- main kernel: one wave = 16 batch rows through the full recurrence ----------
__global__ void __launch_bounds__(32 * WAVES)
seqpred_main(const float* __restrict__ x, const float* __restrict__ noise,
             const _Float16* __restrict__ wsB, const float* __restrict__ wsF,
             float* __restrict__ out)
{
  __shared__ __align__(32) _Float16 scr_all[WAVES][16 * SCR_STRIDE];
  const int lane = threadIdx.x & 31;
  const int wid  = threadIdx.x >> 5;
  const int col  = lane & 15, hs = lane >> 4;
  const int rowBase = (blockIdx.x * WAVES + wid) * 16;
  _Float16* scr = scr_all[wid];

  // noise (constant per row): build A-fragments once (K=64 -> 2 chunks)
  v16h noiseA[2];
  {
    const float* nrow = noise + (size_t)(rowBase + col) * 64;
#pragma unroll
    for (int c = 0; c < 2; ++c) {
      AFrag a;
#pragma unroll
      for (int i = 0; i < 8; ++i) a.e[i]     = (_Float16)nrow[c * 32 + 8 * hs + i];
#pragma unroll
      for (int i = 0; i < 8; ++i) a.e[8 + i] = (_Float16)nrow[c * 32 + 8 * hs + 16 + i];
      noiseA[c] = a.v;
    }
  }

  float c_p[16], c_v[16];
#pragma unroll
  for (int i = 0; i < 16; ++i) { c_p[i] = 0.0f; c_v[i] = 0.0f; }
  v16h hA_p = {};
  v16h hA_v = {};

  float2 x0_[8], xm4_[8], last_[8];

  // ---- encoder: 8 steps over x ----
  const float2* xr = (const float2*)x + (size_t)(rowBase + col) * T_IN;
  for (int t = 0; t < T_IN; ++t) {
    const float2 mine = xr[t];          // lane loads its row; redistribute via wave32 shuffle
    float2 xt[8];
#pragma unroll
    for (int r = 0; r < 8; ++r) {
      xt[r].x = __shfl(mine.x, 8 * hs + r, 32);
      xt[r].y = __shfl(mine.y, 8 * hs + r, 32);
    }
    if (t == 0)        {
#pragma unroll
      for (int r = 0; r < 8; ++r) x0_[r]  = xt[r];
    }
    if (t == T_IN - 4) {
#pragma unroll
      for (int r = 0; r < 8; ++r) xm4_[r] = xt[r];
    }
    if (t == T_IN - 1) {
#pragma unroll
      for (int r = 0; r < 8; ++r) last_[r] = xt[r];
    }
    lstm_cell(xt, c_p, hA_p, wsB, WHHP_OFF / 512, wsF + WIHP_F, wsF + BP_F, scr, lane);
  }

  // ---- autoregressive decode ----
  for (int step = 0; step < OUT_LEN; ++step) {
    if (step > 0)  // uniform branch, EXEC stays full around WMMA
      lstm_cell(last_, c_p, hA_p, wsB, WHHP_OFF / 512, wsF + WIHP_F, wsF + BP_F, scr, lane);
    head_step(step, out, rowBase, hA_p, hA_v, c_v, noiseA,
              last_, x0_, xm4_, wsB, wsF, scr, lane);
  }
}

// ---------- weight prep: fp32 -> f16 WMMA B-fragment layout ----------
__device__ __forceinline__ void pack_B(const float* __restrict__ W, int N, int K,
                                       _Float16* __restrict__ dst, int nTiles, int kChunks,
                                       int tid, int nth)
{
  const int total = kChunks * nTiles * 512;
  for (int i = tid; i < total; i += nth) {
    const int frag = i >> 9;
    const int rem  = i & 511;
    const int lane = rem >> 4;
    const int e    = rem & 15;
    const int c = frag / nTiles, t = frag - c * nTiles;
    const int n = t * 16 + (lane & 15);
    const int k = c * 32 + 8 * (lane >> 4) + (e & 7) + 16 * (e >> 3);   // B layout: lanes 0-15 K {0..7,16..23}, lanes 16-31 K {8..15,24..31}
    const float v = (n < N && k < K) ? W[n * K + k] : 0.0f;
    dst[i] = (_Float16)v;
  }
}

__global__ void seqpred_prep(
    const float* Wih_p, const float* Whh_p, const float* bih_p, const float* bhh_p,
    const float* Wih_v, const float* Whh_v, const float* bih_v, const float* bhh_v,
    const float* W1, const float* b1, const float* W2, const float* b2,
    const float* W3, const float* b3, const float* W4, const float* b4,
    const float* W5, const float* b5,
    _Float16* wsB, float* wsF)
{
  const int tid = blockIdx.x * blockDim.x + threadIdx.x;
  const int nth = gridDim.x * blockDim.x;
  pack_B(Whh_p, 128, 32,  wsB + WHHP_OFF, 8, 1, tid, nth);
  pack_B(Whh_v, 128, 32,  wsB + WHHV_OFF, 8, 1, tid, nth);
  pack_B(W1,    64,  128, wsB + W1_OFF,   4, 4, tid, nth);
  pack_B(W2,    64,  64,  wsB + W2_OFF,   4, 2, tid, nth);
  pack_B(W3,    64,  64,  wsB + W3_OFF,   4, 2, tid, nth);
  pack_B(W4,    64,  64,  wsB + W4_OFF,   4, 2, tid, nth);
  pack_B(W5,    2,   64,  wsB + W5_OFF,   1, 2, tid, nth);
  for (int i = tid; i < 256; i += nth) wsF[WIHP_F + i] = Wih_p[i];
  for (int i = tid; i < 128; i += nth) wsF[BP_F   + i] = bih_p[i] + bhh_p[i];
  for (int i = tid; i < 256; i += nth) wsF[WIHV_F + i] = Wih_v[i];
  for (int i = tid; i < 128; i += nth) wsF[BV_F   + i] = bih_v[i] + bhh_v[i];
  for (int i = tid; i < 64;  i += nth) wsF[B1_F + i] = b1[i];
  for (int i = tid; i < 64;  i += nth) wsF[B2_F + i] = b2[i];
  for (int i = tid; i < 64;  i += nth) wsF[B3_F + i] = b3[i];
  for (int i = tid; i < 64;  i += nth) wsF[B4_F + i] = b4[i];
  for (int i = tid; i < 2;   i += nth) wsF[B5_F + i] = b5[i];
}

extern "C" void kernel_launch(void* const* d_in, const int* in_sizes, int n_in,
                              void* d_out, int out_size, void* d_ws, size_t ws_size,
                              hipStream_t stream) {
  const float* x      = (const float*)d_in[0];
  const float* noise  = (const float*)d_in[1];
  // d_in[2] = out_len (fixed at 12)
  const float* Wih_p  = (const float*)d_in[3];
  const float* Whh_p  = (const float*)d_in[4];
  const float* bih_p  = (const float*)d_in[5];
  const float* bhh_p  = (const float*)d_in[6];
  const float* Wih_v  = (const float*)d_in[7];
  const float* Whh_v  = (const float*)d_in[8];
  const float* bih_v  = (const float*)d_in[9];
  const float* bhh_v  = (const float*)d_in[10];
  const float* W1     = (const float*)d_in[11];
  const float* b1     = (const float*)d_in[12];
  const float* W2     = (const float*)d_in[13];
  const float* b2     = (const float*)d_in[14];
  const float* W3     = (const float*)d_in[15];
  const float* b3     = (const float*)d_in[16];
  const float* W4     = (const float*)d_in[17];
  const float* b4     = (const float*)d_in[18];
  const float* W5     = (const float*)d_in[19];
  const float* b5     = (const float*)d_in[20];

  _Float16* wsB = (_Float16*)d_ws;
  float*    wsF = (float*)((char*)d_ws + (size_t)WSB_HALVES * 2);

  seqpred_prep<<<64, 256, 0, stream>>>(Wih_p, Whh_p, bih_p, bhh_p,
                                       Wih_v, Whh_v, bih_v, bhh_v,
                                       W1, b1, W2, b2, W3, b3, W4, b4, W5, b5,
                                       wsB, wsF);

  const int rows = in_sizes[0] / (T_IN * 2);       // 65536
  const int grid = rows / (WAVES * 16);            // 1024 blocks of 4 waves
  seqpred_main<<<grid, 32 * WAVES, 0, stream>>>(x, noise, wsB, wsF, (float*)d_out);
}
